// RainbowNet_39565238731279
// MI455X (gfx1250) — matrix operand order repaired and checked by reading, not compiled, not run on hardware
//
#include <hip/hip_runtime.h>
#include <cstdint>
#include <cstddef>

typedef _Float16 h16;
typedef __attribute__((ext_vector_type(16))) _Float16 v16h;
typedef __attribute__((ext_vector_type(8)))  _Float16 v8h;
typedef __attribute__((ext_vector_type(8)))  float    v8f;

#define WMMA16(a, b, c) __builtin_amdgcn_wmma_f32_16x16x32_f16( \
    false, (a), false, (b), (short)0, (c), false, false)

// ---------------------------------------------------------------------------
// constants for token building
// ---------------------------------------------------------------------------
__device__ __constant__ int c_cum[10]  = {0, 16, 24, 28, 32, 36, 44, 52, 60, 68};
__device__ __constant__ int c_base[9]  = {59, 124, 157, 174, 191, 208, 241, 274, 307};

// ---------------------------------------------------------------------------
// Weight pre-pack: f32 (K x N, row-major) -> f16 in WMMA B-fragment order.
// Tile (nt, kt): 32 lanes x 16 halves contiguous.
//   lane<16 : K = kt*32 + (h<8 ? h : h+8)     (h = 0..15), N = nt*16 + lane
//   lane>=16: K = kt*32 + 8 + (h<8 ? h : h+8),             N = nt*16 + (lane-16)
// ---------------------------------------------------------------------------
__global__ void pack_w(const float* __restrict__ W, h16* __restrict__ out,
                       int K, int N, int KT, int NTiles) {
  int idx = blockIdx.x * 256 + threadIdx.x;
  int total = NTiles * KT * 512;
  if (idx >= total) return;
  int h    = idx & 15;
  int lane = (idx >> 4) & 31;
  int rem  = idx >> 9;
  int kt   = rem % KT;
  int nt   = rem / KT;
  int k = kt * 32 + ((lane & 16) ? 8 : 0) + (h < 8 ? h : h + 8);
  int n = nt * 16 + (lane & 15);
  float v = (k < K && n < N) ? W[(size_t)k * N + n] : 0.f;
  out[idx] = (h16)v;
}

// ---------------------------------------------------------------------------
// A-fragment load (16-bit A 16x32 layout):
//   lane<16 : halves 0..7 = K k0..k0+7,   halves 8..15 = K k0+16..k0+23
//   lane>=16: halves 0..7 = K k0+8..+15,  halves 8..15 = K k0+24..+31
// base already includes row offset and the +8-half selector for lanes>=16.
// ---------------------------------------------------------------------------
static __device__ __forceinline__ v16h load_a_frag(const h16* base, int kb) {
  v8h lo = *reinterpret_cast<const v8h*>(base + kb);
  v8h hi = *reinterpret_cast<const v8h*>(base + kb + 16);
  v16h a;
#pragma unroll
  for (int i = 0; i < 8; ++i) { a[i] = lo[i]; a[i + 8] = hi[i]; }
  return a;
}

// ---------------------------------------------------------------------------
// GEMM, block = 32 rows x full N (N = NT*128). 8 waves, wave w owns n-slice
// [w*NT*16, ...). Epilogue: +bias, optional LayerNorm(g,beta), ReLU, f16 out.
// A: M x K f16 row-major (lda == K, K multiple of 32). Bp: packed weights.
// ---------------------------------------------------------------------------
template <int NT, bool LN>
__global__ __launch_bounds__(256) void gemm_k1(
    const h16* __restrict__ A, const h16* __restrict__ Bp,
    const float* __restrict__ bias, const float* __restrict__ g,
    const float* __restrict__ beta, h16* __restrict__ out, int K, int N) {
  __shared__ float redS[32][8], redQ[32][8], mrow[32], rrow[32];
  const int lane = threadIdx.x & 31;
  const int wave = threadIdx.x >> 5;
  const int m0   = blockIdx.x * 32;
  const int KT   = K >> 5;
  const int rOff = (lane & 16) ? 8 : 0;
  const h16* a0 = A + (size_t)(m0 + (lane & 15)) * K + ((lane & 16) ? 8 : 0);
  const h16* a1 = a0 + (size_t)16 * K;

  const v8f vzero = {0.f, 0.f, 0.f, 0.f, 0.f, 0.f, 0.f, 0.f};
  v8f acc[2][NT];
#pragma unroll
  for (int mt = 0; mt < 2; ++mt)
#pragma unroll
    for (int nt = 0; nt < NT; ++nt) acc[mt][nt] = vzero;

  for (int kt = 0; kt < KT; ++kt) {
    int kb = kt << 5;
    v16h af0 = load_a_frag(a0, kb);
    v16h af1 = load_a_frag(a1, kb);
#pragma unroll
    for (int nt = 0; nt < NT; ++nt) {
      size_t tile = (size_t)(wave * NT + nt);
      v16h b = *reinterpret_cast<const v16h*>(
          Bp + ((tile * KT + kt) * 32 + lane) * 16);
      acc[0][nt] = WMMA16(af0, b, acc[0][nt]);
      acc[1][nt] = WMMA16(af1, b, acc[1][nt]);
    }
  }
  // bias
#pragma unroll
  for (int nt = 0; nt < NT; ++nt) {
    int n = (wave * NT + nt) * 16 + (lane & 15);
    float bv = bias[n];
#pragma unroll
    for (int v = 0; v < 8; ++v) { acc[0][nt][v] += bv; acc[1][nt][v] += bv; }
  }

  if (LN) {
    // per-row stats: lane-half reduce (16 lanes), then cross-wave via LDS
#pragma unroll
    for (int mt = 0; mt < 2; ++mt)
#pragma unroll
      for (int v = 0; v < 8; ++v) {
        float s = 0.f, q = 0.f;
#pragma unroll
        for (int nt = 0; nt < NT; ++nt) {
          float x = acc[mt][nt][v];
          s += x; q += x * x;
        }
        for (int off = 1; off < 16; off <<= 1) {
          s += __shfl_xor(s, off, 32);
          q += __shfl_xor(q, off, 32);
        }
        int row = mt * 16 + v + rOff;
        if ((lane & 15) == 0) { redS[row][wave] = s; redQ[row][wave] = q; }
      }
    __syncthreads();
    if (threadIdx.x < 32) {
      float s = 0.f, q = 0.f;
      for (int w = 0; w < 8; ++w) { s += redS[threadIdx.x][w]; q += redQ[threadIdx.x][w]; }
      float mean = s / (float)N;
      float var  = q / (float)N - mean * mean;
      mrow[threadIdx.x] = mean;
      rrow[threadIdx.x] = rsqrtf(var + 1e-5f);
    }
    __syncthreads();
#pragma unroll
    for (int mt = 0; mt < 2; ++mt)
#pragma unroll
      for (int v = 0; v < 8; ++v) {
        int row = mt * 16 + v + rOff;
        float m = mrow[row], r = rrow[row];
#pragma unroll
        for (int nt = 0; nt < NT; ++nt) {
          int n = (wave * NT + nt) * 16 + (lane & 15);
          float val = (acc[mt][nt][v] - m) * r * g[n] + beta[n];
          val = fmaxf(val, 0.f);
          out[(size_t)(m0 + row) * N + n] = (h16)val;
        }
      }
  } else {
#pragma unroll
    for (int mt = 0; mt < 2; ++mt)
#pragma unroll
      for (int v = 0; v < 8; ++v) {
        int row = mt * 16 + v + rOff;
#pragma unroll
        for (int nt = 0; nt < NT; ++nt) {
          int n = (wave * NT + nt) * 16 + (lane & 15);
          float val = fmaxf(acc[mt][nt][v], 0.f);
          out[(size_t)(m0 + row) * N + n] = (h16)val;
        }
      }
  }
}

// ---------------------------------------------------------------------------
// Output GEMM: block = 32 rows, 8 waves x 1 n-tile each; grid.y tiles N.
// f32 output with column guard (for N not multiple of 16), bias fused.
// ---------------------------------------------------------------------------
__global__ __launch_bounds__(256) void gemm_k2(
    const h16* __restrict__ A, const h16* __restrict__ Bp,
    const float* __restrict__ bias, float* __restrict__ out,
    int K, int NTpacked, int Ntrue, int ldout) {
  const int lane = threadIdx.x & 31;
  const int wave = threadIdx.x >> 5;
  const int m0   = blockIdx.x * 32;
  const int tile = blockIdx.y * 8 + wave;
  if (tile >= NTpacked) return;
  const int KT = K >> 5;
  const h16* a0 = A + (size_t)(m0 + (lane & 15)) * K + ((lane & 16) ? 8 : 0);
  const h16* a1 = a0 + (size_t)16 * K;
  const v8f vzero = {0.f, 0.f, 0.f, 0.f, 0.f, 0.f, 0.f, 0.f};
  v8f acc0 = vzero, acc1 = vzero;
  for (int kt = 0; kt < KT; ++kt) {
    int kb = kt << 5;
    v16h af0 = load_a_frag(a0, kb);
    v16h af1 = load_a_frag(a1, kb);
    v16h b = *reinterpret_cast<const v16h*>(
        Bp + (((size_t)tile * KT + kt) * 32 + lane) * 16);
    acc0 = WMMA16(af0, b, acc0);
    acc1 = WMMA16(af1, b, acc1);
  }
  int n = tile * 16 + (lane & 15);
  if (n < Ntrue) {
    float bv = bias[n];
    int rOff = (lane & 16) ? 8 : 0;
#pragma unroll
    for (int v = 0; v < 8; ++v) {
      out[(size_t)(m0 + v + rOff) * ldout + n]      = acc0[v] + bv;
      out[(size_t)(m0 + 16 + v + rOff) * ldout + n] = acc1[v] + bv;
    }
  }
}

// ---------------------------------------------------------------------------
// attn_prep: q = Wq cls + bq (constant across batch!), wvec[h] = q_h^T Wk,
// cb[h] = q_h . bk_h.  One block of 128 threads.
// ---------------------------------------------------------------------------
__global__ __launch_bounds__(128) void attn_prep(
    const float* __restrict__ cls, const float* __restrict__ ipW,
    const float* __restrict__ ipb, float* __restrict__ wvec,
    float* __restrict__ cbv) {
  __shared__ float qv[128];
  int r = threadIdx.x;
  float q = ipb[r];
  const float* wr = ipW + (size_t)r * 128;
  for (int c = 0; c < 128; ++c) q += wr[c] * cls[c];
  qv[r] = q;
  __syncthreads();
#pragma unroll
  for (int h = 0; h < 4; ++h) {
    float a = 0.f;
    for (int u = 0; u < 32; ++u)
      a += qv[h * 32 + u] * ipW[(size_t)(128 + h * 32 + u) * 128 + r];
    wvec[h * 128 + r] = a;
  }
  if (r < 4) {
    float a = 0.f;
    for (int u = 0; u < 32; ++u) a += qv[r * 32 + u] * ipb[128 + r * 32 + u];
    cbv[r] = a;
  }
}

// ---------------------------------------------------------------------------
// build_h0: per-sample tokens -> embed+LN -> collapsed attention -> pooled,
// writes h0 row = [state(1360) | pooled(128) | 0-pad(16)] as f16 (stride 1504)
// ---------------------------------------------------------------------------
__global__ __launch_bounds__(128) void build_h0(
    const float* __restrict__ state, const float* __restrict__ eW,
    const float* __restrict__ eb, const float* __restrict__ lg,
    const float* __restrict__ lb, const float* __restrict__ cls,
    const float* __restrict__ wvec, const float* __restrict__ cbv,
    const float* __restrict__ ipW, const float* __restrict__ ipb,
    const float* __restrict__ opW, const float* __restrict__ opb,
    h16* __restrict__ h0) {
  __shared__ float xs[69 * 128];
  __shared__ float sc[4 * 69];
  __shared__ float at[4 * 69];
  __shared__ int   msk[69];
  __shared__ float ybuf[4 * 128];
  __shared__ float obuf[128];
  __shared__ float pbuf[128];
  const int tid = threadIdx.x, lane = tid & 31, wave = tid >> 5;
  const int i = blockIdx.x;
  const float* st = state + (size_t)i * 1360;
  if (tid == 0) msk[0] = 0;

  for (int j = wave; j < 69; j += 4) {
    const int c4 = lane * 4;
    if (j == 0) {
#pragma unroll
      for (int t = 0; t < 4; ++t) xs[c4 + t] = cls[c4 + t];
      continue;
    }
    int s = j - 1;
    int c = 0;
    while (s >= c_cum[c + 1]) ++c;
    int slot = s - c_cum[c];
    int off  = c_base[c] + 1 + slot * 4;
    int lo = 1020 + off, po = 680 + off;
    float present = st[lo], dx = st[lo + 1], dy = st[lo + 2], dist = st[lo + 3];
    float pp = st[po], pdx = st[po + 1], pdy = st[po + 2];
    float vv = (present > 0.5f && pp > 0.5f) ? 1.f : 0.f;
    float tok[7] = {dx, dy, dist, (dx - pdx) * vv, (dy - pdy) * vv,
                    (float)c * 0.125f, present};
    if (lane == 0) msk[j] = (present < 0.5f) ? 1 : 0;
    float e[4];
#pragma unroll
    for (int t = 0; t < 4; ++t) {
      float a = eb[c4 + t];
#pragma unroll
      for (int f = 0; f < 7; ++f) a += tok[f] * eW[f * 128 + c4 + t];
      e[t] = a;
    }
    float s1 = e[0] + e[1] + e[2] + e[3];
    for (int o = 1; o < 32; o <<= 1) s1 += __shfl_xor(s1, o, 32);
    float mean = s1 * (1.f / 128.f);
    float s2 = 0.f;
#pragma unroll
    for (int t = 0; t < 4; ++t) { float d = e[t] - mean; s2 += d * d; }
    for (int o = 1; o < 32; o <<= 1) s2 += __shfl_xor(s2, o, 32);
    float rstd = rsqrtf(s2 * (1.f / 128.f) + 1e-5f);
#pragma unroll
    for (int t = 0; t < 4; ++t)
      xs[j * 128 + c4 + t] = (e[t] - mean) * rstd * lg[c4 + t] + lb[c4 + t];
  }
  __syncthreads();
  if (tid == 0) {  // all-empty handling (reference semantics)
    int all = 1;
    for (int j = 1; j < 69; ++j) if (!msk[j]) { all = 0; break; }
    if (all) for (int j = 1; j < 69; ++j) msk[j] = 0;
  }
  __syncthreads();
  const float isc = 0.17677669529663687f;  // 1/sqrt(32)
  for (int idx = tid; idx < 4 * 69; idx += 128) {
    int h = idx / 69, j = idx % 69;
    const float* wv = wvec + h * 128;
    const float* x  = xs + j * 128;
    float sacc = cbv[h];
    for (int cc = 0; cc < 128; ++cc) sacc += wv[cc] * x[cc];
    sacc *= isc;
    if (msk[j]) sacc = -1.0e9f;
    sc[h * 69 + j] = sacc;
  }
  __syncthreads();
  if (tid < 4) {
    float mx = -3.4e38f;
    for (int j = 0; j < 69; ++j) mx = fmaxf(mx, sc[tid * 69 + j]);
    float sum = 0.f;
    for (int j = 0; j < 69; ++j) {
      float e2 = expf(sc[tid * 69 + j] - mx);
      at[tid * 69 + j] = e2; sum += e2;
    }
    float inv = 1.f / sum;
    for (int j = 0; j < 69; ++j) at[tid * 69 + j] *= inv;
  }
  __syncthreads();
  {  // y^h = sum_j a^h_j x_j
    int cc = tid;
#pragma unroll
    for (int h = 0; h < 4; ++h) {
      float y = 0.f;
      for (int j = 0; j < 69; ++j) y += at[h * 69 + j] * xs[j * 128 + cc];
      ybuf[h * 128 + cc] = y;
    }
  }
  __syncthreads();
  {  // out = Wv y + bv (row r uses head r/32)
    int r = tid, h = r >> 5;
    float o = ipb[256 + r];
    const float* wr = ipW + (size_t)(256 + r) * 128;
    const float* y  = ybuf + h * 128;
    for (int cc = 0; cc < 128; ++cc) o += wr[cc] * y[cc];
    obuf[r] = o;
  }
  __syncthreads();
  {  // pooled = Wo out + bo
    int p = tid;
    float o = opb[p];
    const float* wr = opW + (size_t)p * 128;
    for (int r = 0; r < 128; ++r) o += wr[r] * obuf[r];
    pbuf[p] = o;
  }
  __syncthreads();
  h16* hr = h0 + (size_t)i * 1504;
  for (int t = tid; t < 1504; t += 128) {
    float v;
    if (t < 1360)      v = st[t];
    else if (t < 1488) v = pbuf[t - 1360];
    else               v = 0.f;
    hr[t] = (h16)v;
  }
}

// ---------------------------------------------------------------------------
// finalize: q = val + adv - mean_a(adv); softmax over atoms; in-place on out
// ---------------------------------------------------------------------------
__global__ __launch_bounds__(128) void finalize(float* __restrict__ out,
                                                const float* __restrict__ val) {
  __shared__ float adv[4131];
  __shared__ float vl[51];
  __shared__ float mn[51];
  int tid = threadIdx.x, i = blockIdx.x;
  float* orow = out + (size_t)i * 4131;
  for (int t = tid; t < 4131; t += 128) adv[t] = orow[t];
  if (tid < 51) vl[tid] = val[(size_t)i * 51 + tid];
  __syncthreads();
  if (tid < 51) {
    float s = 0.f;
    for (int a = 0; a < 81; ++a) s += adv[a * 51 + tid];
    mn[tid] = s * (1.f / 81.f);
  }
  __syncthreads();
  if (tid < 81) {
    int base = tid * 51;
    float l[51];
    float mx = -3.4e38f;
    for (int k = 0; k < 51; ++k) {
      float q = vl[k] + adv[base + k] - mn[k];
      l[k] = q; mx = fmaxf(mx, q);
    }
    float s = 0.f;
    for (int k = 0; k < 51; ++k) { float e = expf(l[k] - mx); l[k] = e; s += e; }
    float inv = 1.f / s;
    for (int k = 0; k < 51; ++k) orow[base + k] = l[k] * inv;
  }
}

// ---------------------------------------------------------------------------
extern "C" void kernel_launch(void* const* d_in, const int* in_sizes, int n_in,
                              void* d_out, int out_size, void* d_ws,
                              size_t ws_size, hipStream_t stream) {
  (void)in_sizes; (void)n_in; (void)out_size; (void)ws_size;
  const float* state = (const float*)d_in[0];
  const float* eW    = (const float*)d_in[1];
  const float* ebv   = (const float*)d_in[2];
  const float* lg    = (const float*)d_in[3];
  const float* lbv   = (const float*)d_in[4];
  const float* cls   = (const float*)d_in[5];
  const float* ipW   = (const float*)d_in[6];
  const float* ipb   = (const float*)d_in[7];
  const float* opW   = (const float*)d_in[8];
  const float* opb   = (const float*)d_in[9];
  const float* t0W   = (const float*)d_in[10];
  const float* t0b   = (const float*)d_in[11];
  const float* t0g   = (const float*)d_in[12];
  const float* t0be  = (const float*)d_in[13];
  const float* t1W   = (const float*)d_in[14];
  const float* t1b   = (const float*)d_in[15];
  const float* t1g   = (const float*)d_in[16];
  const float* t1be  = (const float*)d_in[17];
  const float* vfW   = (const float*)d_in[18];
  const float* vfb   = (const float*)d_in[19];
  const float* voW   = (const float*)d_in[20];
  const float* vob   = (const float*)d_in[21];
  const float* afW   = (const float*)d_in[22];
  const float* afb   = (const float*)d_in[23];
  const float* aoW   = (const float*)d_in[24];
  const float* aob   = (const float*)d_in[25];
  float* dout = (float*)d_out;

  char* ws = (char*)d_ws;
  size_t o = 0;
  auto take = [&](size_t b) { size_t r = o; o += (b + 255) & ~(size_t)255; return r; };
  h16* packW0  = (h16*)(ws + take((size_t)64 * 47 * 512 * 2));
  h16* packW1  = (h16*)(ws + take((size_t)64 * 32 * 512 * 2));
  h16* packWvf = (h16*)(ws + take((size_t)32 * 32 * 512 * 2));
  h16* packWaf = (h16*)(ws + take((size_t)32 * 32 * 512 * 2));
  h16* packWvo = (h16*)(ws + take((size_t)4 * 16 * 512 * 2));
  h16* packWao = (h16*)(ws + take((size_t)259 * 16 * 512 * 2));
  float* wvec  = (float*)(ws + take(512 * 4));
  float* cbv   = (float*)(ws + take(4 * 4));
  h16* h0 = (h16*)(ws + take((size_t)8192 * 1504 * 2));
  h16* h1 = (h16*)(ws + take((size_t)8192 * 1024 * 2));
  h16* h2 = (h16*)(ws + take((size_t)8192 * 1024 * 2));
  h16* vh = (h16*)(ws + take((size_t)8192 * 512 * 2));
  h16* ah = (h16*)(ws + take((size_t)8192 * 512 * 2));
  float* valb = (float*)(ws + take((size_t)8192 * 51 * 4));

  auto packGrid = [](int nt, int kt) { return (nt * kt * 512 + 255) / 256; };
  pack_w<<<packGrid(64, 47), 256, 0, stream>>>(t0W, packW0, 1488, 1024, 47, 64);
  pack_w<<<packGrid(64, 32), 256, 0, stream>>>(t1W, packW1, 1024, 1024, 32, 64);
  pack_w<<<packGrid(32, 32), 256, 0, stream>>>(vfW, packWvf, 1024, 512, 32, 32);
  pack_w<<<packGrid(32, 32), 256, 0, stream>>>(afW, packWaf, 1024, 512, 32, 32);
  pack_w<<<packGrid(4, 16), 256, 0, stream>>>(voW, packWvo, 512, 51, 16, 4);
  pack_w<<<packGrid(259, 16), 256, 0, stream>>>(aoW, packWao, 512, 4131, 16, 259);

  attn_prep<<<1, 128, 0, stream>>>(cls, ipW, ipb, wvec, cbv);
  build_h0<<<8192, 128, 0, stream>>>(state, eW, ebv, lg, lbv, cls, wvec, cbv,
                                     ipW, ipb, opW, opb, h0);

  gemm_k1<8, true><<<256, 256, 0, stream>>>(h0, packW0, t0b, t0g, t0be, h1, 1504, 1024);
  gemm_k1<8, true><<<256, 256, 0, stream>>>(h1, packW1, t1b, t1g, t1be, h2, 1024, 1024);
  gemm_k1<4, false><<<256, 256, 0, stream>>>(h2, packWvf, vfb, nullptr, nullptr, vh, 1024, 512);
  gemm_k1<4, false><<<256, 256, 0, stream>>>(h2, packWaf, afb, nullptr, nullptr, ah, 1024, 512);

  gemm_k2<<<dim3(256, 1), 256, 0, stream>>>(vh, packWvo, vob, valb, 512, 4, 51, 51);
  gemm_k2<<<dim3(256, 33), 256, 0, stream>>>(ah, packWao, aob, dout, 512, 259, 4131, 4131);

  finalize<<<8192, 128, 0, stream>>>(dout, valb);
}
